// MultiHeadAttention_27178553049803
// MI455X (gfx1250) — matrix operand order repaired and checked by reading, not compiled, not run on hardware
//
#include <hip/hip_runtime.h>
#include <stdint.h>

// MultiHeadAttention forward for MI455X (gfx1250, wave32, WMMA f16->f32).
// B=2, S=2048, D=2048, H=16, Hd=128.

#define D_MODEL 2048
#define N_HEADS 16
#define HEAD_DIM 128
#define BATCH 2
#define SEQ 2048
#define M_ROWS (BATCH * SEQ)   // 4096

typedef _Float16 half_t;
typedef __attribute__((ext_vector_type(16))) _Float16 v16h;
typedef __attribute__((ext_vector_type(8)))  _Float16 v8h;
typedef __attribute__((ext_vector_type(4)))  _Float16 v4h;
typedef __attribute__((ext_vector_type(8)))  float    v8f;
typedef __attribute__((ext_vector_type(4)))  float    v4f;

__device__ __forceinline__ v8f zero8() {
  v8f z;
#pragma unroll
  for (int i = 0; i < 8; ++i) z[i] = 0.0f;
  return z;
}

__device__ __forceinline__ v8f wmma_f16(v16h a, v16h b, v8f c) {
  // (neg_a, A, neg_b, B, c_mod, C, reuse_a, reuse_b)
  return __builtin_amdgcn_wmma_f32_16x16x32_f16(false, a, false, b, (short)0, c,
                                                false, false);
}

// ---- gfx1250 async global->LDS copy (ASYNCcnt-tracked). lds = wave-relative
// LDS byte address (low 32 bits of the generic shared pointer); 16B per lane.
template <int OFF>
__device__ __forceinline__ void async_ld_b128(uint32_t lds, const void* g) {
  asm volatile("global_load_async_to_lds_b128 %0, %1, off offset:%2"
               :: "v"(lds), "v"(g), "n"(OFF) : "memory");
}
__device__ __forceinline__ void wait_async_le8() {
  asm volatile("s_wait_asynccnt 0x8" ::: "memory");
}
__device__ __forceinline__ void wait_async_0() {
  asm volatile("s_wait_asynccnt 0x0" ::: "memory");
}
__device__ __forceinline__ uint32_t lds_off32(const void* p) {
  return (uint32_t)(size_t)p;   // LDS aperture: addr[31:0] = LDS offset
}

// A-fragment 16x32 f16 (M x K), row-major source with leading dim `ld`.
// Contiguous k order used here (B matches): lanes 0-15 hold M=lane, K=[k0,k0+16);
//             lanes 16-31 hold M=lane-16, K=[k0+16,k0+32).
__device__ __forceinline__ v16h load_fragA_g(const half_t* __restrict__ src,
                                             int ld, int row0, int k0, int lane) {
  const int m  = lane & 15;
  const int kb = k0 + ((lane >> 4) << 4);
  const half_t* p = src + (size_t)(row0 + m) * ld + kb;
  v8h c0 = *(const v8h*)(p);
  v8h c1 = *(const v8h*)(p + 8);
  v16h f;
#pragma unroll
  for (int i = 0; i < 8; ++i) { f[i] = c0[i]; f[i + 8] = c1[i]; }
  return f;
}

// Same A-fragment layout from an LDS tile.
__device__ __forceinline__ v16h load_fragA_lds(const half_t* src, int ld,
                                               int k0, int lane) {
  const int m  = lane & 15;
  const int kb = k0 + ((lane >> 4) << 4);
  const half_t* p = src + m * ld + kb;
  v8h c0 = *(const v8h*)(p);
  v8h c1 = *(const v8h*)(p + 8);
  v16h f;
#pragma unroll
  for (int i = 0; i < 8; ++i) { f[i] = c0[i]; f[i + 8] = c1[i]; }
  return f;
}

// B-fragment 32x16 f16 (K x N), src[(col)*ld + k] = B[k][col].
// Contiguous k order (as A): lanes 0-15 hold N=lane, K=[k0,k0+16); lanes 16-31 K=[k0+16,k0+32).
__device__ __forceinline__ v16h load_fragB_g(const half_t* __restrict__ src,
                                             int ld, int col0, int k0, int lane) {
  const int n  = lane & 15;
  const int kb = k0 + ((lane >> 4) << 4);
  return *(const v16h*)(src + (size_t)(col0 + n) * ld + kb);
}
__device__ __forceinline__ v16h load_fragB_lds(const half_t* src, int ld,
                                               int col0, int k0, int lane) {
  const int n  = lane & 15;
  const int kb = k0 + ((lane >> 4) << 4);
  return *(const v16h*)(src + (size_t)(col0 + n) * ld + kb);
}

// ---------------------------------------------------------------- convert
__global__ void cvt_f32_f16_kernel(const float* __restrict__ src,
                                   half_t* __restrict__ dst, int n4) {
  int i = blockIdx.x * blockDim.x + threadIdx.x;
  if (i < n4) {
    v4f v = *(const v4f*)(src + 4 * (size_t)i);
    v4h o;
#pragma unroll
    for (int j = 0; j < 4; ++j) o[j] = (half_t)v[j];
    *(v4h*)(dst + 4 * (size_t)i) = o;
  }
}

// ------------------------------------------------- Q/K/V projection GEMM
// Y = X @ W^T + b. Block tile 128x128, 8 waves (2 M x 4 N), wave tile 64x32.
// mode 0: Q -> [b,h,s,d] scaled by 1/sqrt(Hd); 1: K -> [b,h,s,d];
// mode 2: V -> [b,h,d,s] (transposed store, one v8h per fragment).
__global__ __launch_bounds__(256, 1)
void proj_qkv_kernel(const half_t* __restrict__ X, const half_t* __restrict__ W,
                     const float* __restrict__ bias, half_t* __restrict__ dst,
                     int mode, float qscale) {
  const int lane = threadIdx.x & 31;
  const int wave = threadIdx.x >> 5;
  const int mw = wave & 1;
  const int nw = wave >> 1;
  const int row0 = blockIdx.y * 128 + mw * 64;
  const int col0 = blockIdx.x * 128 + nw * 32;

  v8f acc[4][2];
#pragma unroll
  for (int i = 0; i < 4; ++i)
#pragma unroll
    for (int j = 0; j < 2; ++j) acc[i][j] = zero8();

  for (int k0 = 0; k0 < D_MODEL; k0 += 32) {
    // Speculative prefetch two k-steps ahead (TH=0: tail overshoot is
    // silently dropped per ISA; keeps the loop branch-free).
    __builtin_prefetch(X + (size_t)(row0 + lane) * D_MODEL + k0 + 64, 0, 3);
    __builtin_prefetch(X + (size_t)(row0 + 32 + lane) * D_MODEL + k0 + 64, 0, 3);
    __builtin_prefetch(W + (size_t)(col0 + lane) * D_MODEL + k0 + 64, 0, 3);
    v16h a[4];
#pragma unroll
    for (int i = 0; i < 4; ++i)
      a[i] = load_fragA_g(X, D_MODEL, row0 + 16 * i, k0, lane);
#pragma unroll
    for (int j = 0; j < 2; ++j) {
      v16h b = load_fragB_g(W, D_MODEL, col0 + 16 * j, k0, lane);
#pragma unroll
      for (int i = 0; i < 4; ++i) acc[i][j] = wmma_f16(a[i], b, acc[i][j]);
    }
  }

  const int bidx = row0 / SEQ;          // tile lies in one batch
  const int srow = row0 % SEQ;
  const int lgrp = lane >> 4;
#pragma unroll
  for (int i = 0; i < 4; ++i) {
#pragma unroll
    for (int j = 0; j < 2; ++j) {
      const int gcol = col0 + 16 * j + (lane & 15);
      const int h = gcol >> 7;
      const int d = gcol & (HEAD_DIM - 1);
      const float bv = bias[gcol];
      if (mode == 2) {
        v8h pack;
#pragma unroll
        for (int r = 0; r < 8; ++r) pack[r] = (half_t)(acc[i][j][r] + bv);
        const int s0 = srow + 16 * i + lgrp * 8;
        half_t* p = dst + (((size_t)bidx * N_HEADS + h) * HEAD_DIM + d) * SEQ + s0;
        *(v8h*)p = pack;
      } else {
#pragma unroll
        for (int r = 0; r < 8; ++r) {
          const int s = srow + 16 * i + lgrp * 8 + r;
          float v = acc[i][j][r] + bv;
          if (mode == 0) v *= qscale;
          dst[(((size_t)bidx * N_HEADS + h) * SEQ + s) * HEAD_DIM + d] = (half_t)v;
        }
      }
    }
  }
}

// ----------------------------------------------------- output projection
__global__ __launch_bounds__(256, 1)
void proj_out_kernel(const half_t* __restrict__ X, const half_t* __restrict__ W,
                     const float* __restrict__ bias, float* __restrict__ out) {
  const int lane = threadIdx.x & 31;
  const int wave = threadIdx.x >> 5;
  const int mw = wave & 1;
  const int nw = wave >> 1;
  const int row0 = blockIdx.y * 128 + mw * 64;
  const int col0 = blockIdx.x * 128 + nw * 32;

  v8f acc[4][2];
#pragma unroll
  for (int i = 0; i < 4; ++i)
#pragma unroll
    for (int j = 0; j < 2; ++j) acc[i][j] = zero8();

  for (int k0 = 0; k0 < D_MODEL; k0 += 32) {
    __builtin_prefetch(X + (size_t)(row0 + lane) * D_MODEL + k0 + 64, 0, 3);
    __builtin_prefetch(X + (size_t)(row0 + 32 + lane) * D_MODEL + k0 + 64, 0, 3);
    __builtin_prefetch(W + (size_t)(col0 + lane) * D_MODEL + k0 + 64, 0, 3);
    v16h a[4];
#pragma unroll
    for (int i = 0; i < 4; ++i)
      a[i] = load_fragA_g(X, D_MODEL, row0 + 16 * i, k0, lane);
#pragma unroll
    for (int j = 0; j < 2; ++j) {
      v16h b = load_fragB_g(W, D_MODEL, col0 + 16 * j, k0, lane);
#pragma unroll
      for (int i = 0; i < 4; ++i) acc[i][j] = wmma_f16(a[i], b, acc[i][j]);
    }
  }

  const int lgrp = lane >> 4;
#pragma unroll
  for (int i = 0; i < 4; ++i) {
#pragma unroll
    for (int j = 0; j < 2; ++j) {
      const int gcol = col0 + 16 * j + (lane & 15);
      const float bv = bias[gcol];
#pragma unroll
      for (int r = 0; r < 8; ++r) {
        const int row = row0 + 16 * i + lgrp * 8 + r;
        out[(size_t)row * D_MODEL + gcol] = acc[i][j][r] + bv;
      }
    }
  }
}

// -------------------------------------------------------- flash attention
// Grid: (SEQ/128, B*H), 8 waves/block, wave owns 16 q-rows x full Hd=128.
// K/Vt tiles double-buffered in LDS via global_load_async_to_lds_b128
// (8 async instrs per wave per tile batch -> s_wait_asynccnt 8 pipelining).
// Dynamic LDS: K 2x16KB | V 2x16KB | P 8x2KB = 80KB.
#define KT_ELEMS (64 * 128)
#define VT_ELEMS (128 * 64)

__global__ __launch_bounds__(256, 1)
void attn_kernel(const half_t* __restrict__ Q, const half_t* __restrict__ K,
                 const half_t* __restrict__ Vt, half_t* __restrict__ ctx) {
  extern __shared__ char smem[];
  half_t* Ktile = (half_t*)smem;                 // [2][64*128]
  half_t* Vtile = Ktile + 2 * KT_ELEMS;          // [2][128*64]
  half_t* Ptile = Vtile + 2 * VT_ELEMS;          // [8][16*64]

  const int tid = threadIdx.x;
  const int lane = tid & 31;
  const int wave = tid >> 5;
  const int bh = blockIdx.y;
  const int b = bh >> 4;         // / N_HEADS
  const int h = bh & 15;
  const int q0 = blockIdx.x * 128 + wave * 16;
  const int lgrp = lane >> 4;

  const half_t* Qb = Q + (size_t)bh * SEQ * HEAD_DIM;
  const char* Kg = (const char*)(K + (size_t)bh * SEQ * HEAD_DIM);
  const char* Vg = (const char*)(Vt + (size_t)bh * HEAD_DIM * SEQ);

  // Q fragments for this wave's 16 rows, full Hd=128 (4 x K=32 chunks).
  v16h aq[4];
#pragma unroll
  for (int t = 0; t < 4; ++t) aq[t] = load_fragA_g(Qb, HEAD_DIM, q0, 32 * t, lane);

  v8f acc[8];
#pragma unroll
  for (int dt = 0; dt < 8; ++dt) acc[dt] = zero8();
  float rm[8], rl[8];
#pragma unroll
  for (int r = 0; r < 8; ++r) { rm[r] = -__builtin_inff(); rl[r] = 0.0f; }

  half_t* pl = Ptile + wave * (16 * 64);   // wave-private P tile

  // Async stage of one 64-wide k-tile (8 async b128 per thread = per wave).
  auto issue_tile = [&](int it) {
    const int k0 = it * 64;
    const int buf = it & 1;
    {   // K tile: 64 rows x 256B, fully contiguous 16KB chunk of K.
      uint32_t l = lds_off32(Ktile + buf * KT_ELEMS) + tid * 16;
      const char* g = Kg + (size_t)k0 * (HEAD_DIM * 2) + tid * 16;
      async_ld_b128<0>(l, g);
      async_ld_b128<4096>(l, g);
      async_ld_b128<8192>(l, g);
      async_ld_b128<12288>(l, g);
    }
    {   // Vt tile: 128 rows x 128B, row stride SEQ*2 in global.
      const int row = tid >> 1;
      const int col = (tid & 1) * 16;
      uint32_t l = lds_off32(Vtile + buf * VT_ELEMS) + row * 128 + col;
      const char* g = Vg + (size_t)row * (SEQ * 2) + (size_t)k0 * 2 + col;
      async_ld_b128<0>(l, g);
      async_ld_b128<32>(l, g);
      async_ld_b128<64>(l, g);
      async_ld_b128<96>(l, g);
    }
  };

  constexpr int NIT = SEQ / 64;
  issue_tile(0);

  for (int it = 0; it < NIT; ++it) {
    const int buf = it & 1;
    if (it + 1 < NIT) { issue_tile(it + 1); wait_async_le8(); }
    else              { wait_async_0(); }
    __syncthreads();   // all waves' tile portions landed in LDS

    const half_t* Kt = Ktile + buf * KT_ELEMS;   // [s][d], ld=128
    const half_t* Vb = Vtile + buf * VT_ELEMS;   // [d][s], ld=64

    // S = Q @ K^T (scale folded into Q).
    v8f sc[4];
#pragma unroll
    for (int nt = 0; nt < 4; ++nt) sc[nt] = zero8();
#pragma unroll
    for (int nt = 0; nt < 4; ++nt) {
#pragma unroll
      for (int kt = 0; kt < 4; ++kt) {
        v16h bk = load_fragB_lds(Kt, HEAD_DIM, 16 * nt, 32 * kt, lane);
        sc[nt] = wmma_f16(aq[kt], bk, sc[nt]);
      }
    }

    // Online softmax; row r lives across the 16 lanes of this half-wave.
    float corr[8], ps[8];
#pragma unroll
    for (int r = 0; r < 8; ++r) {
      float mx = rm[r];
#pragma unroll
      for (int nt = 0; nt < 4; ++nt) mx = fmaxf(mx, sc[nt][r]);
#pragma unroll
      for (int off = 1; off < 16; off <<= 1) mx = fmaxf(mx, __shfl_xor(mx, off, 32));
      corr[r] = __expf(rm[r] - mx);   // exp(-inf)=0 on first tile
      rm[r] = mx;
      ps[r] = 0.0f;
    }

    // P = exp(S - m) -> LDS row-major [16][64] for re-fragmenting.
#pragma unroll
    for (int nt = 0; nt < 4; ++nt) {
#pragma unroll
      for (int r = 0; r < 8; ++r) {
        float p = __expf(sc[nt][r] - rm[r]);
        ps[r] += p;
        const int m = (lgrp << 3) + r;
        const int n = nt * 16 + (lane & 15);
        pl[m * 64 + n] = (half_t)p;
      }
    }
#pragma unroll
    for (int r = 0; r < 8; ++r) {
      float s = ps[r];
#pragma unroll
      for (int off = 1; off < 16; off <<= 1) s += __shfl_xor(s, off, 32);
      rl[r] = rl[r] * corr[r] + s;
    }
#pragma unroll
    for (int dt = 0; dt < 8; ++dt)
#pragma unroll
      for (int r = 0; r < 8; ++r) acc[dt][r] *= corr[r];

    // P @ V from LDS.
    v16h pa[2];
    pa[0] = load_fragA_lds(pl, 64, 0, lane);
    pa[1] = load_fragA_lds(pl, 64, 32, lane);
#pragma unroll
    for (int kt = 0; kt < 2; ++kt) {
#pragma unroll
      for (int dt = 0; dt < 8; ++dt) {
        v16h bv = load_fragB_lds(Vb, 64, dt * 16, 32 * kt, lane);
        acc[dt] = wmma_f16(pa[kt], bv, acc[dt]);
      }
    }
    __syncthreads();   // protect buffer reuse by the next async batch
  }

  // Normalize and store ctx[b, q, h*Hd + d] as f16.
#pragma unroll
  for (int dt = 0; dt < 8; ++dt) {
    const int d = dt * 16 + (lane & 15);
#pragma unroll
    for (int r = 0; r < 8; ++r) {
      const int qrow = q0 + (lgrp << 3) + r;
      float o = acc[dt][r] / rl[r];
      ctx[((size_t)b * SEQ + qrow) * D_MODEL + h * HEAD_DIM + d] = (half_t)o;
    }
  }
}

// ------------------------------------------------------------------ launch
extern "C" void kernel_launch(void* const* d_in, const int* in_sizes, int n_in,
                              void* d_out, int out_size, void* d_ws, size_t ws_size,
                              hipStream_t stream) {
  (void)in_sizes; (void)n_in; (void)out_size; (void)ws_size;
  const float* q  = (const float*)d_in[0];
  const float* k  = (const float*)d_in[1];
  const float* v  = (const float*)d_in[2];
  const float* Wq = (const float*)d_in[3];
  const float* bq = (const float*)d_in[4];
  const float* Wk = (const float*)d_in[5];
  const float* bk = (const float*)d_in[6];
  const float* Wv = (const float*)d_in[7];
  const float* bv = (const float*)d_in[8];
  const float* Wo = (const float*)d_in[9];
  const float* bo = (const float*)d_in[10];
  float* out = (float*)d_out;

  const size_t XN = (size_t)M_ROWS * D_MODEL;   // 8,388,608 halves
  const size_t WN = (size_t)D_MODEL * D_MODEL;  // 4,194,304 halves
  half_t* ws  = (half_t*)d_ws;
  half_t* Xq  = ws;  ws += XN;
  half_t* Xk  = ws;  ws += XN;
  half_t* Xv  = ws;  ws += XN;
  half_t* Wqh = ws;  ws += WN;
  half_t* Wkh = ws;  ws += WN;
  half_t* Wvh = ws;  ws += WN;
  half_t* Woh = ws;  ws += WN;
  half_t* Qh  = ws;  ws += XN;
  half_t* Kh  = ws;  ws += XN;
  half_t* Vth = ws;  ws += XN;
  half_t* Ctx = ws;  ws += XN;   // total ~144 MiB

  const int thr = 256;
  {
    int n4 = (int)(XN / 4);
    dim3 g((n4 + thr - 1) / thr);
    cvt_f32_f16_kernel<<<g, thr, 0, stream>>>(q, Xq, n4);
    cvt_f32_f16_kernel<<<g, thr, 0, stream>>>(k, Xk, n4);
    cvt_f32_f16_kernel<<<g, thr, 0, stream>>>(v, Xv, n4);
  }
  {
    int n4 = (int)(WN / 4);
    dim3 g((n4 + thr - 1) / thr);
    cvt_f32_f16_kernel<<<g, thr, 0, stream>>>(Wq, Wqh, n4);
    cvt_f32_f16_kernel<<<g, thr, 0, stream>>>(Wk, Wkh, n4);
    cvt_f32_f16_kernel<<<g, thr, 0, stream>>>(Wv, Wvh, n4);
    cvt_f32_f16_kernel<<<g, thr, 0, stream>>>(Wo, Woh, n4);
  }

  const dim3 gg(D_MODEL / 128, M_ROWS / 128);   // (16, 32)
  const float qscale = 0.08838834764831845f;    // 1/sqrt(128)
  proj_qkv_kernel<<<gg, thr, 0, stream>>>(Xq, Wqh, bq, Qh, 0, qscale);
  proj_qkv_kernel<<<gg, thr, 0, stream>>>(Xk, Wkh, bk, Kh, 1, 1.0f);
  proj_qkv_kernel<<<gg, thr, 0, stream>>>(Xv, Wvh, bv, Vth, 2, 1.0f);

  const size_t attn_lds = (2 * KT_ELEMS + 2 * VT_ELEMS + 8 * 16 * 64) * sizeof(half_t); // 80 KB
  attn_kernel<<<dim3(SEQ / 128, BATCH * N_HEADS), thr, attn_lds, stream>>>(Qh, Kh, Vth, Ctx);

  proj_out_kernel<<<gg, thr, 0, stream>>>(Ctx, Woh, bo, out);
}
